// MultiLayerRNN_49297634624142
// MI455X (gfx1250) — matrix operand order repaired and checked by reading, not compiled, not run on hardware
//
#include <hip/hip_runtime.h>
#include <hip/hip_bf16.h>

// MultiLayerRNN on gfx1250 (MI455X):
//  - xp GEMMs + recurrent scan use v_wmma_f32_16x16x32_bf16 (fp32 accumulate)
//  - recurrence is batch-parallel: 4 workgroups x 16 batch rows, NO cross-WG sync
//  - w_hh fully on-WGP: 8/16 K-steps in VGPRs (64 regs), 8/16 in LDS (270 KB)
//  - activations in [T, B, H] so per-step XP/YS traffic is contiguous and
//    epilogue addressing is base + constant-immediate (saves VGPRs)

#define B_ 64
#define T_ 1024
#define I_ 64
#define H_ 512
#define L_ 3

#define KK_REG 8                     // w_hh K-steps resident in VGPRs
#define BL_K   (512 - KK_REG * 32)   // = 256 k-columns kept in LDS
#define BL_S   (BL_K + 8)            // padded LDS row stride (bf16 elems)

typedef __attribute__((ext_vector_type(16))) __bf16 v16bf;
typedef __attribute__((ext_vector_type(8)))  float  v8f;

union FragBF {
  v16bf v;
  uint4 q[2];
};

// ---------------------------------------------------------------- weight cvt
__global__ __launch_bounds__(256) void cvt_f32_bf16(const float* __restrict__ src,
                                                    __bf16* __restrict__ dst, int n) {
  int i = blockIdx.x * blockDim.x + threadIdx.x;
  if (i < n) dst[i] = (__bf16)src[i];
}

// ---------------------------------------------------------------- xp = A @ W^T + (b_ih + b_hh)
// A: [M, K] (f32 for layer 0, bf16 for layers 1/2), W: [H_, K] bf16 row-major.
// Block = 256 threads (8 waves), 64x64 output tile.
// TRANS_OUT: input rows are b*T+t but XP is written at row t*B+b ([T,B,H]).
template <bool A_IS_F32, bool TRANS_OUT>
__global__ __launch_bounds__(256) void xp_gemm(
    const void* __restrict__ Araw,
    const __bf16* __restrict__ W,
    const float* __restrict__ b_ih,
    const float* __restrict__ b_hh,
    float* __restrict__ XP,
    int K)
{
  const int lane  = threadIdx.x & 31;
  const int wave  = threadIdx.x >> 5;
  const int ntile = wave & 3;
  const int mt0   = (wave >> 2) * 2;

  const int M0 = blockIdx.x * 64;
  const int N0 = blockIdx.y * 64 + ntile * 16;

  const int hi = lane >> 4;   // half-wave select (ISA fragment layout)
  const int lr = lane & 15;   // row (A) / col (B,C) within tile
  const int n  = N0 + lr;

  v8f c0 = {}; v8f c1 = {};
  const int ksteps = K >> 5;

  for (int kk = 0; kk < ksteps; ++kk) {
    const int kbA = (kk << 5) + hi * 8;   // A frag: elems 0..7 @kbA, 8..15 @kbA+16
    const int kbB = (kk << 5) + hi * 8;   // B frag: elems 0..7 @kbB, 8..15 @kbB+16

    FragBF b;
    {
      const __bf16* wp = W + (size_t)n * K + kbB;
      b.q[0] = *(const uint4*)(wp);
      b.q[1] = *(const uint4*)(wp + 16);
      if (kk + 1 < ksteps) __builtin_prefetch(wp + 32, 0, 0);
    }

    FragBF a0, a1;
    if (A_IS_F32) {
      const float* A  = (const float*)Araw;
      const float* p0 = A + (size_t)(M0 + (mt0 + 0) * 16 + lr) * K + kbA;
      const float* p1 = A + (size_t)(M0 + (mt0 + 1) * 16 + lr) * K + kbA;
      #pragma unroll
      for (int j = 0; j < 8; ++j) {
        a0.v[j]     = (__bf16)p0[j];
        a0.v[8 + j] = (__bf16)p0[16 + j];
        a1.v[j]     = (__bf16)p1[j];
        a1.v[8 + j] = (__bf16)p1[16 + j];
      }
    } else {
      const __bf16* A  = (const __bf16*)Araw;
      const __bf16* p0 = A + (size_t)(M0 + (mt0 + 0) * 16 + lr) * K + kbA;
      const __bf16* p1 = A + (size_t)(M0 + (mt0 + 1) * 16 + lr) * K + kbA;
      a0.q[0] = *(const uint4*)(p0);
      a0.q[1] = *(const uint4*)(p0 + 16);
      a1.q[0] = *(const uint4*)(p1);
      a1.q[1] = *(const uint4*)(p1 + 16);
    }

    c0 = __builtin_amdgcn_wmma_f32_16x16x32_bf16(false, a0.v, false, b.v, (short)0, c0, false, false);
    c1 = __builtin_amdgcn_wmma_f32_16x16x32_bf16(false, a1.v, false, b.v, (short)0, c1, false, false);
  }

  const float bias = b_ih[n] + b_hh[n];
  #pragma unroll
  for (int r = 0; r < 8; ++r) {
    const int m = r + hi * 8;  // C/D layout: lanes 16-31 hold M = 8 + r
    int row0 = M0 + (mt0 + 0) * 16 + m;
    int row1 = M0 + (mt0 + 1) * 16 + m;
    if (TRANS_OUT) {   // input row b*T+t -> XP row t*B+b
      row0 = (row0 & (T_ - 1)) * B_ + (row0 >> 10);
      row1 = (row1 & (T_ - 1)) * B_ + (row1 >> 10);
    }
    XP[(size_t)row0 * H_ + n] = c0[r] + bias;
    XP[(size_t)row1 * H_ + n] = c1[r] + bias;
  }
}

// ---------------------------------------------------------------- recurrent scan (one layer)
// Grid = 4 blocks (16 batch rows each), block = 1024 threads = 32 waves.
// Wave w owns N-tile [16w, 16w+16). w_hh: KK_REG K-steps in VGPRs + rest in LDS.
// XP / YS are [T, B, H] so per-step global traffic is one contiguous block.
__global__ __launch_bounds__(1024) void rnn_scan(
    const float* __restrict__ XP,    // [T_, B_, H_] precomputed input projection
    const __bf16* __restrict__ Whh,  // [H_, H_] bf16 row-major ([n][k])
    const float* __restrict__ h0,    // [B_, H_] initial hidden for this layer
    __bf16* __restrict__ YS,         // [T_, B_, H_] layer output (bf16)
    float* __restrict__ hT)          // [B_, H_] final hidden (f32, into d_out)
{
  constexpr int HS = H_ + 8;                    // padded h row stride (bf16)
  __shared__ __bf16 hbuf[16 * HS];              // 16.6 KB
  __shared__ __bf16 bLds[H_ * BL_S];            // 512 x 264 bf16 = 270 KB

  const int lane  = threadIdx.x & 31;
  const int wave  = threadIdx.x >> 5;
  const int hi    = lane >> 4;
  const int lr    = lane & 15;
  const int n     = wave * 16 + lr;
  const int bbase = blockIdx.x * 16;

  // Resident recurrent-weight B-fragments for K-steps [0, KK_REG).
  FragBF bfr[KK_REG];
  #pragma unroll
  for (int kk = 0; kk < KK_REG; ++kk) {
    const __bf16* wp = Whh + (size_t)n * H_ + (kk << 5) + hi * 8;
    bfr[kk].q[0] = *(const uint4*)(wp);
    bfr[kk].q[1] = *(const uint4*)(wp + 16);
  }

  // Tail K-steps [KK_REG, 16) of w_hh into LDS (each wave copies its n-rows).
  for (int e = lane; e < 16 * (BL_K / 8); e += 32) {
    const int row = e / (BL_K / 8);          // 0..15 within this wave's n-tile
    const int ch  = e % (BL_K / 8);          // 16-byte chunk within BL_K columns
    const int nn  = wave * 16 + row;
    *(uint4*)(bLds + nn * BL_S + ch * 8) =
        *(const uint4*)(Whh + (size_t)nn * H_ + KK_REG * 32 + ch * 8);
  }

  // h_{-1} into LDS (bf16).
  for (int e = threadIdx.x; e < 16 * H_; e += 1024) {
    const int m = e >> 9, k = e & (H_ - 1);
    hbuf[m * HS + k] = (__bf16)h0[(size_t)(bbase + m) * H_ + k];
  }
  __syncthreads();

  for (int t = 0; t < T_; ++t) {
    v8f c = {};
    #pragma unroll
    for (int kk = 0; kk < KK_REG; ++kk) {
      FragBF a;
      const __bf16* hp = hbuf + lr * HS + (kk << 5) + hi * 8;
      a.q[0] = *(const uint4*)(hp);
      a.q[1] = *(const uint4*)(hp + 16);
      c = __builtin_amdgcn_wmma_f32_16x16x32_bf16(false, a.v, false, bfr[kk].v, (short)0, c, false, false);
    }
    #pragma unroll
    for (int kk = KK_REG; kk < 16; ++kk) {
      FragBF a, b;
      const __bf16* hp = hbuf + lr * HS + (kk << 5) + hi * 8;
      a.q[0] = *(const uint4*)(hp);
      a.q[1] = *(const uint4*)(hp + 16);
      const __bf16* bp = bLds + n * BL_S + ((kk - KK_REG) << 5) + hi * 8;
      b.q[0] = *(const uint4*)(bp);
      b.q[1] = *(const uint4*)(bp + 16);
      c = __builtin_amdgcn_wmma_f32_16x16x32_bf16(false, a.v, false, b.v, (short)0, c, false, false);
    }

    // xp add + tanh; [T,B,H] layout: per-r address = base + r*H_*4 (immediate)
    const float* xr = XP + (size_t)t * B_ * H_ + (size_t)(bbase + hi * 8) * H_ + n;
    #pragma unroll
    for (int r = 0; r < 8; ++r)
      c[r] = tanhf(c[r] + xr[(size_t)r * H_]);

    __syncthreads();  // all waves finished reading h_{t-1}

    __bf16* yr = YS + (size_t)t * B_ * H_ + (size_t)(bbase + hi * 8) * H_ + n;
    #pragma unroll
    for (int r = 0; r < 8; ++r) {
      const __bf16 hb = (__bf16)c[r];
      hbuf[(r + hi * 8) * HS + n] = hb;
      yr[(size_t)r * H_] = hb;
    }

    __syncthreads();  // h_t visible before next step's reads
  }

  // Final hidden state (f32) from LDS.
  for (int e = threadIdx.x; e < 16 * H_; e += 1024) {
    const int m = e >> 9, k = e & (H_ - 1);
    hT[(size_t)(bbase + m) * H_ + k] = (float)hbuf[m * HS + k];
  }
}

// ---------------------------------------------------------------- linear head (O = 1)
// YS rows are t*B+b; out is [B, T].
__global__ __launch_bounds__(256) void fc_head(
    const __bf16* __restrict__ YS, const float* __restrict__ fc_w,
    const float* __restrict__ fc_b, float* __restrict__ out)
{
  const int gid = blockIdx.x * blockDim.x + threadIdx.x;  // row index t*B+b
  if (gid >= B_ * T_) return;
  const __bf16* y = YS + (size_t)gid * H_;
  float acc = 0.f;
  for (int h = 0; h < H_; h += 8) {
    union { uint4 q; __bf16 e[8]; } u;
    u.q = *(const uint4*)(y + h);
    #pragma unroll
    for (int j = 0; j < 8; ++j) acc += (float)u.e[j] * fc_w[h + j];
  }
  const int b = gid & (B_ - 1);
  const int t = gid >> 6;
  out[(size_t)b * T_ + t] = acc + fc_b[0];
}

// ---------------------------------------------------------------- launch
extern "C" void kernel_launch(void* const* d_in, const int* in_sizes, int n_in,
                              void* d_out, int out_size, void* d_ws, size_t ws_size,
                              hipStream_t stream) {
  const float* input  = (const float*)d_in[0];
  const float* hidden = (const float*)d_in[1];
  const float* w_ih[3] = {(const float*)d_in[2], (const float*)d_in[6],  (const float*)d_in[10]};
  const float* w_hh[3] = {(const float*)d_in[3], (const float*)d_in[7],  (const float*)d_in[11]};
  const float* b_ih[3] = {(const float*)d_in[4], (const float*)d_in[8],  (const float*)d_in[12]};
  const float* b_hh[3] = {(const float*)d_in[5], (const float*)d_in[9],  (const float*)d_in[13]};
  const float* fc_w = (const float*)d_in[14];
  const float* fc_b = (const float*)d_in[15];

  // Workspace: XP f32 (128MB) | YS bf16 (64MB) | bf16 weight copies (~2.7MB)
  char* ws = (char*)d_ws;
  float*  XP = (float*)ws;
  __bf16* YS = (__bf16*)(ws + (size_t)B_ * T_ * H_ * 4);
  char* wp = ws + (size_t)B_ * T_ * H_ * 4 + (size_t)B_ * T_ * H_ * 2;
  __bf16* wihb[3]; __bf16* whhb[3];
  wihb[0] = (__bf16*)wp; wp += (size_t)H_ * I_ * 2;
  wihb[1] = (__bf16*)wp; wp += (size_t)H_ * H_ * 2;
  wihb[2] = (__bf16*)wp; wp += (size_t)H_ * H_ * 2;
  whhb[0] = (__bf16*)wp; wp += (size_t)H_ * H_ * 2;
  whhb[1] = (__bf16*)wp; wp += (size_t)H_ * H_ * 2;
  whhb[2] = (__bf16*)wp; wp += (size_t)H_ * H_ * 2;

  cvt_f32_bf16<<<(H_ * I_ + 255) / 256, 256, 0, stream>>>(w_ih[0], wihb[0], H_ * I_);
  cvt_f32_bf16<<<(H_ * H_ + 255) / 256, 256, 0, stream>>>(w_ih[1], wihb[1], H_ * H_);
  cvt_f32_bf16<<<(H_ * H_ + 255) / 256, 256, 0, stream>>>(w_ih[2], wihb[2], H_ * H_);
  cvt_f32_bf16<<<(H_ * H_ + 255) / 256, 256, 0, stream>>>(w_hh[0], whhb[0], H_ * H_);
  cvt_f32_bf16<<<(H_ * H_ + 255) / 256, 256, 0, stream>>>(w_hh[1], whhb[1], H_ * H_);
  cvt_f32_bf16<<<(H_ * H_ + 255) / 256, 256, 0, stream>>>(w_hh[2], whhb[2], H_ * H_);

  float* out_seq = (float*)d_out;                 // [B, T, 1]
  float* hT_base = (float*)d_out + B_ * T_;       // [L, B, H]

  const dim3 gemm_grid(B_ * T_ / 64, H_ / 64);
  for (int l = 0; l < 3; ++l) {
    if (l == 0)   // layer 0 reads f32 input ([B,T,I] rows) and writes XP as [T,B,H]
      xp_gemm<true , true ><<<gemm_grid, 256, 0, stream>>>(input, wihb[0], b_ih[0], b_hh[0], XP, I_);
    else          // layers 1/2 read YS ([T,B,H] rows) and keep that order
      xp_gemm<false, false><<<gemm_grid, 256, 0, stream>>>(YS,    wihb[l], b_ih[l], b_hh[l], XP, H_);
    rnn_scan<<<4, 1024, 0, stream>>>(XP, whhb[l], hidden + (size_t)l * B_ * H_,
                                     YS, hT_base + (size_t)l * B_ * H_);
  }
  fc_head<<<(B_ * T_ + 255) / 256, 256, 0, stream>>>(YS, fc_w, fc_b, out_seq);
}
